// AttentionGRUDecoder_5798205849781
// MI455X (gfx1250) — matrix operand order repaired and checked by reading, not compiled, not run on hardware
//
#include <hip/hip_runtime.h>
#include <hip/hip_bf16.h>
#include <cstdint>

// Problem constants (from reference): B=128, T=150, H=512, A=512, V=600
#define BB   128
#define TT   150
#define HH   512
#define AA   512
#define S2A  1024      // 2*A
#define VV   600
#define KA   1112      // V + H  (= V + A as well)
#define KAP  1120      // KA padded to multiple of 32
#define G3H  1536      // 3*H
#define VP   608       // V padded to multiple of 16
#define OUTLD (TT*VV)  // 90000, row stride of output_seq per batch element
#define LDSPAD 8       // +8 halves per LDS B row: dword stride 564 -> 16 distinct banks

typedef __attribute__((ext_vector_type(16))) _Float16 v16h;
typedef __attribute__((ext_vector_type(8)))  float    v8f;
typedef __attribute__((ext_vector_type(4)))  int      v4i;

union FragAB { v16h h; unsigned int u[8]; };
union FragC  { v8f  f; float e[8]; };
union H2U    { unsigned int u; _Float16 h[2]; };

typedef __attribute__((address_space(1))) v4i GV4;   // global int4
typedef __attribute__((address_space(3))) v4i LV4;   // LDS int4

// 16-byte global->LDS copy; async (ASYNCcnt-tracked, no VGPR landing) when the
// gfx1250 builtin is available, plain load/store otherwise.
// AS-pointer construction goes through integers: global keeps the full VA;
// for LDS the generic pointer's low 32 bits equal the LDS offset (ISA 10.2
// aperture rule), so truncation yields the correct AS(3) address.
__device__ __forceinline__ void copy16_g2l(const _Float16* g, _Float16* l)
{
#if __has_builtin(__builtin_amdgcn_global_load_async_to_lds_b128)
    __builtin_amdgcn_global_load_async_to_lds_b128(
        (GV4*)(unsigned long long)(uintptr_t)g,
        (LV4*)(unsigned int)(uintptr_t)l, 0, 0);
#else
    *(float4*)l = *(const float4*)g;
#endif
}

__device__ __forceinline__ void wait_async_then_barrier()
{
#if __has_builtin(__builtin_amdgcn_global_load_async_to_lds_b128)
#if __has_builtin(__builtin_amdgcn_s_wait_asynccnt)
    __builtin_amdgcn_s_wait_asynccnt(0);
#else
    asm volatile("s_wait_asynccnt 0x0" ::: "memory");
#endif
#endif
    __syncthreads();
}

// ---------------------------------------------------------------------------
// WMMA GEMM: C[M=128, N] = A16[M, Kp] @ W16[N, Kp]^T + bias, optional relu,
// optional f32 and/or f16 outputs. Block = 256 threads = 8 waves; wave w owns
// the 16-row M-subtile, block owns one 16-col N tile (gridDim.x = N/16).
// The 16xKp weight tile is staged once into LDS with async-to-LDS loads and
// shared by all 8 waves; A fragments come straight from (L2-hot) global.
// Fragment layouts per ISA 7.12.2.
// ---------------------------------------------------------------------------
__global__ void __launch_bounds__(256)
wmma_gemm_kernel(const _Float16* __restrict__ A, int lda,
                 const _Float16* __restrict__ W, int ldw,
                 const float* __restrict__ bias,
                 float* __restrict__ outF, _Float16* __restrict__ outH,
                 int ldo, int nLimit, int Kp, int applyRelu)
{
    __shared__ __align__(16) _Float16 Btile[16 * (KAP + LDSPAD)];

    const int wave  = threadIdx.x >> 5;   // 0..7 -> M subtile
    const int lane  = threadIdx.x & 31;
    const int grp   = lane >> 4;          // lane group 0/1
    const int l16   = lane & 15;
    const int mBase = wave * 16;
    const int nBase = blockIdx.x * 16;
    const int ldsK  = Kp + LDSPAD;

    // ---- cooperative async stage of the 16xKp B tile into LDS
    {
        const int kv = Kp >> 3;           // 16-byte vectors per row
        for (int i = threadIdx.x; i < 16 * kv; i += 256) {
            const int row = i / kv;
            const int col = (i - row * kv) << 3;
            copy16_g2l(W + (size_t)(nBase + row) * ldw + col,
                       &Btile[row * ldsK + col]);
        }
    }
    wait_async_then_barrier();

    const _Float16* aRow = A + (size_t)(mBase + l16) * lda;
    const _Float16* bRow = &Btile[l16 * ldsK];

    FragC acc{};   // zero f32 accumulator (8 VGPRs)

    for (int k0 = 0; k0 < Kp; k0 += 32) {
        if (k0 + 32 < Kp)
            __builtin_prefetch(aRow + k0 + 32, 0, 3);   // global_prefetch_b8
        FragAB fa, fb;
#pragma unroll
        for (int v = 0; v < 8; ++v) {
            // A 16x32 f16: vgpr v -> seg=v/4, vin=v%4; k = seg*16 + grp*8 + vin*2
            const int ka = k0 + ((v >> 2) << 4) + (grp << 3) + ((v & 3) << 1);
            fa.u[v] = *(const unsigned int*)(aRow + ka);
            // B 32x16 f16: same k order as A; k = seg*16 + grp*8 + vin*2
            const int kb = k0 + ((v >> 2) << 4) + (grp << 3) + ((v & 3) << 1);
            fb.u[v] = *(const unsigned int*)(bRow + kb);   // ds_load from LDS
        }
        acc.f = __builtin_amdgcn_wmma_f32_16x16x32_f16(
            /*neg_a=*/false, fa.h, /*neg_b=*/false, fb.h,
            /*c_mod=*/(short)0, acc.f, /*reuse_a=*/false, /*reuse_b=*/false);
    }

    const int n = nBase + l16;
    const float bval = (n < nLimit) ? bias[n] : 0.0f;
    if (n < nLimit) {
#pragma unroll
        for (int r = 0; r < 8; ++r) {
            const int m = mBase + (grp << 3) + r;   // C/D: vgpr r -> M = r + grp*8
            float v = acc.e[r] + bval;
            if (applyRelu) v = v > 0.0f ? v : 0.0f;
            if (outF) outF[(size_t)m * ldo + n] = v;
            if (outH) outH[(size_t)m * ldo + n] = (_Float16)v;
        }
    }
}

// ---------------------------------------------------------------------------
// Row softmax over [128, 1024]
// ---------------------------------------------------------------------------
__global__ void __launch_bounds__(256)
softmax_kernel(const float* __restrict__ logits, float* __restrict__ aw)
{
    __shared__ float red[256];
    const int b = blockIdx.x;
    const float* row = logits + (size_t)b * S2A;
    float vals[4];
    float lmax = -1e30f;
#pragma unroll
    for (int i = 0; i < 4; ++i) {
        vals[i] = row[threadIdx.x + i * 256];
        lmax = fmaxf(lmax, vals[i]);
    }
    red[threadIdx.x] = lmax; __syncthreads();
    for (int s = 128; s > 0; s >>= 1) {
        if ((int)threadIdx.x < s) red[threadIdx.x] = fmaxf(red[threadIdx.x], red[threadIdx.x + s]);
        __syncthreads();
    }
    lmax = red[0]; __syncthreads();
    float lsum = 0.0f;
#pragma unroll
    for (int i = 0; i < 4; ++i) { vals[i] = __expf(vals[i] - lmax); lsum += vals[i]; }
    red[threadIdx.x] = lsum; __syncthreads();
    for (int s = 128; s > 0; s >>= 1) {
        if ((int)threadIdx.x < s) red[threadIdx.x] += red[threadIdx.x + s];
        __syncthreads();
    }
    const float inv = 1.0f / red[0];
#pragma unroll
    for (int i = 0; i < 4; ++i) aw[(size_t)b * S2A + threadIdx.x + i * 256] = vals[i] * inv;
}

// ---------------------------------------------------------------------------
// attn[b,a] = sum_s aw[b,s] * enc[b,s,a]   (batched matvec; the HBM-bound
// stream: ~134 MB f16 per step). grid = (B, 8): each block handles a 128-wide
// s-chunk, partial sums combined via global_atomic_add_f32.
// ---------------------------------------------------------------------------
__global__ void __launch_bounds__(256)
einsum_kernel(const float* __restrict__ aw,
              const _Float16* __restrict__ enc16,
              const float* __restrict__ enc32,
              float* __restrict__ attn, int useF16)
{
    __shared__ float sa[128];
    const int b  = blockIdx.x;
    const int s0 = blockIdx.y * 128;
    if (threadIdx.x < 128) sa[threadIdx.x] = aw[(size_t)b * S2A + s0 + threadIdx.x];
    __syncthreads();

    const int a0 = threadIdx.x * 2;
    float acc0 = 0.0f, acc1 = 0.0f;
    if (useF16) {
        const _Float16* base = enc16 + ((size_t)b * S2A + s0) * AA + a0;
        for (int s = 0; s < 128; ++s) {
            H2U p; p.u = *(const unsigned int*)(base + (size_t)s * AA);
            const float w = sa[s];
            acc0 += w * (float)p.h[0];
            acc1 += w * (float)p.h[1];
        }
    } else {
        const float* base = enc32 + ((size_t)b * S2A + s0) * AA + a0;
        for (int s = 0; s < 128; ++s) {
            const float2 v = *(const float2*)(base + (size_t)s * AA);
            const float w = sa[s];
            acc0 += w * v.x;
            acc1 += w * v.y;
        }
    }
    atomicAdd(&attn[(size_t)b * AA + a0], acc0);
    atomicAdd(&attn[(size_t)b * AA + a0 + 1], acc1);
}

// ---------------------------------------------------------------------------
// GRU gate math: h_new = (1-z)*n + z*h ; writes hF (f32) and f16 h into the
// h-slot (cols 600..1111) of temp16 for next step's GEMMs.
// ---------------------------------------------------------------------------
__global__ void __launch_bounds__(256)
gate_kernel(const float* __restrict__ gi, const float* __restrict__ gh,
            float* __restrict__ hF, _Float16* __restrict__ temp16)
{
    const int idx = blockIdx.x * 256 + threadIdx.x;   // 128*512 = 65536
    const int b = idx >> 9, j = idx & 511;
    const float* gib = gi + (size_t)b * G3H;
    const float* ghb = gh + (size_t)b * G3H;
    const float r = 1.0f / (1.0f + __expf(-(gib[j]        + ghb[j])));
    const float z = 1.0f / (1.0f + __expf(-(gib[512 + j]  + ghb[512 + j])));
    const float n = tanhf(gib[1024 + j] + r * ghb[1024 + j]);
    const float h = hF[idx];
    const float hn = (1.0f - z) * n + z * h;
    hF[idx] = hn;
    temp16[(size_t)b * KAP + VV + j] = (_Float16)hn;
}

// decoder input (teacher forcing) -> f16 cols 0..599 of temp16 and cat216
__global__ void __launch_bounds__(256)
buildx_kernel(const float* __restrict__ tgt, _Float16* __restrict__ temp16,
              _Float16* __restrict__ cat216, int t)
{
    const int b = blockIdx.x;
    for (int v = threadIdx.x; v < VV; v += 256) {
        const float x = (t == 0) ? 0.0f : tgt[((size_t)b * TT + (t - 1)) * VV + v];
        const _Float16 hx = (_Float16)x;
        temp16[(size_t)b * KAP + v] = hx;
        cat216[(size_t)b * KAP + v] = hx;
    }
}

// attn -> f16 cols 600..1111 of cat216
__global__ void __launch_bounds__(256)
cat2_kernel(const float* __restrict__ attn, _Float16* __restrict__ cat216)
{
    const int idx = blockIdx.x * 256 + threadIdx.x;   // 65536
    const int b = idx >> 9, j = idx & 511;
    cat216[(size_t)b * KAP + VV + j] = (_Float16)attn[idx];
}

__global__ void __launch_bounds__(256)
zero_f32_kernel(float* __restrict__ p, int n)
{
    const int idx = blockIdx.x * 256 + threadIdx.x;
    if (idx < n) p[idx] = 0.0f;
}

// zero the K padding cols (1112..1119) once per call
__global__ void __launch_bounds__(256)
padinit_kernel(_Float16* __restrict__ temp16, _Float16* __restrict__ cat216)
{
    const int idx = blockIdx.x * 256 + threadIdx.x;   // 128*8 = 1024
    if (idx < BB * (KAP - KA)) {
        const int b = idx >> 3, j = KA + (idx & 7);
        temp16[(size_t)b * KAP + j] = (_Float16)0.0f;
        cat216[(size_t)b * KAP + j] = (_Float16)0.0f;
    }
}

// h0 = hidden[-1] -> f32 buffer + f16 h-slot of temp16
__global__ void __launch_bounds__(256)
inith_kernel(const float* __restrict__ hid, float* __restrict__ hF,
             _Float16* __restrict__ temp16)
{
    const int idx = blockIdx.x * 256 + threadIdx.x;   // 65536
    const int b = idx >> 9, j = idx & 511;
    const float v = hid[idx];
    hF[idx] = v;
    temp16[(size_t)b * KAP + VV + j] = (_Float16)v;
}

// weight f32[N,K] -> f16[Np,Kp] with zero padding
__global__ void __launch_bounds__(256)
cvtw_kernel(const float* __restrict__ src, _Float16* __restrict__ dst,
            int Nsrc, int K, int Np, int Kp)
{
    const int total = Np * Kp;
    for (int idx = blockIdx.x * 256 + threadIdx.x; idx < total; idx += gridDim.x * 256) {
        const int n = idx / Kp, k = idx - n * Kp;
        dst[idx] = (n < Nsrc && k < K) ? (_Float16)src[(size_t)n * K + k] : (_Float16)0.0f;
    }
}

// encoder_outputs f32 -> f16 (one-time; halves the 40GB streaming volume)
__global__ void __launch_bounds__(256)
cvtenc_kernel(const float* __restrict__ src, _Float16* __restrict__ dst)
{
    const size_t pairs = (size_t)BB * S2A * AA / 2;
    for (size_t i = blockIdx.x * 256 + threadIdx.x; i < pairs; i += (size_t)gridDim.x * 256) {
        const float2 v = *(const float2*)(src + i * 2);
        H2U p; p.h[0] = (_Float16)v.x; p.h[1] = (_Float16)v.y;
        *(unsigned int*)(dst + i * 2) = p.u;
    }
}

// final hidden -> tail of d_out
__global__ void __launch_bounds__(256)
copyh_kernel(const float* __restrict__ hF, float* __restrict__ out)
{
    const int idx = blockIdx.x * 256 + threadIdx.x;   // 65536
    out[idx] = hF[idx];
}

// ---------------------------------------------------------------------------
extern "C" void kernel_launch(void* const* d_in, const int* in_sizes, int n_in,
                              void* d_out, int out_size, void* d_ws, size_t ws_size,
                              hipStream_t stream)
{
    const float* enc = (const float*)d_in[0];
    const float* hid = (const float*)d_in[1];
    const float* tgt = (const float*)d_in[2];
    // d_in[3] = max_length (device scalar; T fixed at 150)
    const float* Wa  = (const float*)d_in[4];
    const float* ba  = (const float*)d_in[5];
    const float* Wc  = (const float*)d_in[6];
    const float* bc  = (const float*)d_in[7];
    const float* Wih = (const float*)d_in[8];
    const float* Whh = (const float*)d_in[9];
    const float* bih = (const float*)d_in[10];
    const float* bhh = (const float*)d_in[11];
    const float* Wo  = (const float*)d_in[12];
    const float* bo  = (const float*)d_in[13];
    float* out = (float*)d_out;

    // ---- workspace carve-up (256B aligned); enc16 last so the small state
    // fits even if ws can't hold the f16 encoder copy.
    char* ws = (char*)d_ws;
    size_t off = 0;
    auto take = [&](size_t bytes) -> char* {
        char* p = ws + off;
        off = (off + bytes + 255) & ~(size_t)255;
        return p;
    };
    _Float16* temp16  = (_Float16*)take((size_t)BB * KAP * 2);
    _Float16* cat216  = (_Float16*)take((size_t)BB * KAP * 2);
    _Float16* Wa16    = (_Float16*)take((size_t)S2A * KAP * 2);
    _Float16* Wc16    = (_Float16*)take((size_t)HH  * KAP * 2);
    _Float16* Wih16   = (_Float16*)take((size_t)G3H * HH  * 2);
    _Float16* Whh16   = (_Float16*)take((size_t)G3H * HH  * 2);
    _Float16* Wo16    = (_Float16*)take((size_t)VP  * HH  * 2);
    _Float16* gruin16 = (_Float16*)take((size_t)BB  * HH  * 2);
    float* logits = (float*)take((size_t)BB * S2A * 4);
    float* aw     = (float*)take((size_t)BB * S2A * 4);
    float* attn   = (float*)take((size_t)BB * AA  * 4);
    float* gi     = (float*)take((size_t)BB * G3H * 4);
    float* gh     = (float*)take((size_t)BB * G3H * 4);
    float* hF     = (float*)take((size_t)BB * HH  * 4);
    _Float16* enc16 = (_Float16*)take((size_t)BB * S2A * AA * 2);
    const int useF16Enc = (off <= ws_size) ? 1 : 0;

    // ---- one-time (per call) conversions / init
    cvtw_kernel<<<2048, 256, 0, stream>>>(Wa,  Wa16,  S2A, KA, S2A, KAP);
    cvtw_kernel<<<2048, 256, 0, stream>>>(Wc,  Wc16,  HH,  KA, HH,  KAP);
    cvtw_kernel<<<2048, 256, 0, stream>>>(Wih, Wih16, G3H, HH, G3H, HH);
    cvtw_kernel<<<2048, 256, 0, stream>>>(Whh, Whh16, G3H, HH, G3H, HH);
    cvtw_kernel<<<2048, 256, 0, stream>>>(Wo,  Wo16,  VV,  HH, VP,  HH);
    if (useF16Enc) cvtenc_kernel<<<4096, 256, 0, stream>>>(enc, enc16);
    inith_kernel<<<256, 256, 0, stream>>>(hid, hF, temp16);
    padinit_kernel<<<4, 256, 0, stream>>>(temp16, cat216);

    // ---- sequential decoder steps
    for (int t = 0; t < TT; ++t) {
        // decoder input (teacher forcing) into temp16/cat216 cols 0..599
        buildx_kernel<<<BB, 256, 0, stream>>>(tgt, temp16, cat216, t);

        // logits = [x|h] @ Wa^T + ba           [128,1024]
        wmma_gemm_kernel<<<S2A / 16, 256, 0, stream>>>(
            temp16, KAP, Wa16, KAP, ba, logits, nullptr, S2A, S2A, KAP, 0);

        // aw = softmax(logits)
        softmax_kernel<<<BB, 256, 0, stream>>>(logits, aw);

        // attn = sum_s aw[b,s] * enc[b,s,:]
        zero_f32_kernel<<<(BB * AA + 255) / 256, 256, 0, stream>>>(attn, BB * AA);
        einsum_kernel<<<dim3(BB, 8), 256, 0, stream>>>(aw, enc16, enc, attn, useF16Enc);

        // cat2 = [x|attn]
        cat2_kernel<<<(BB * AA + 255) / 256, 256, 0, stream>>>(attn, cat216);

        // gru_in = relu(cat2 @ Wc^T + bc) -> f16
        wmma_gemm_kernel<<<HH / 16, 256, 0, stream>>>(
            cat216, KAP, Wc16, KAP, bc, nullptr, gruin16, HH, HH, KAP, 1);

        // gi = gru_in @ W_ih^T + b_ih ; gh = h @ W_hh^T + b_hh
        wmma_gemm_kernel<<<G3H / 16, 256, 0, stream>>>(
            gruin16, HH, Wih16, HH, bih, gi, nullptr, G3H, G3H, HH, 0);
        wmma_gemm_kernel<<<G3H / 16, 256, 0, stream>>>(
            temp16 + VV, KAP, Whh16, HH, bhh, gh, nullptr, G3H, G3H, HH, 0);

        // h_new (also refreshes f16 h-slot of temp16)
        gate_kernel<<<(BB * HH + 255) / 256, 256, 0, stream>>>(gi, gh, hF, temp16);

        // out_tok = h_new @ Wo^T + bo  -> output_seq[:, t, :]
        wmma_gemm_kernel<<<VP / 16, 256, 0, stream>>>(
            temp16 + VV, KAP, Wo16, HH, bo, out + (size_t)t * VV, nullptr,
            OUTLD, VV, HH, 0);
    }

    // hidden tail of d_out: [1, B, H]
    copyh_kernel<<<256, 256, 0, stream>>>(hF, out + (size_t)BB * TT * VV);
}
